// RocLMAttention_51874615001796
// MI455X (gfx1250) — matrix operand order repaired and checked by reading, not compiled, not run on hardware
//
#include <hip/hip_runtime.h>
#include <hip/hip_bf16.h>

// ---------------------------------------------------------------------------
// Problem constants (from reference)
// ---------------------------------------------------------------------------
constexpr int Bb  = 2;
constexpr int Ss  = 2048;
constexpr int Hh  = 1024;
constexpr int NH  = 16;
constexpr int NKV = 8;
constexpr int Dd  = 128;
constexpr float EPS = 1e-6f;

typedef __attribute__((ext_vector_type(16))) __bf16 v16bf;
typedef __attribute__((ext_vector_type(8)))  float  v8f;

union BFrag  { v16bf v; unsigned short u[16]; uint4 q[2]; };

__device__ __forceinline__ unsigned short f2bf(float f) {
  union { float f; unsigned int u; } c; c.f = f;
  unsigned int u = c.u;
  u += 0x7FFFu + ((u >> 16) & 1u);          // round-to-nearest-even
  return (unsigned short)(u >> 16);
}

#if defined(__gfx1250__)
// Packed f32->bf16 convert: one VALU op instead of ~6 for the manual rounding.
__device__ __forceinline__ unsigned pack2bf(float lo, float hi) {
  unsigned r;
  asm("v_cvt_pk_bf16_f32 %0, %1, %2" : "=v"(r) : "v"(lo), "v"(hi));
  return r;
}
#else
__device__ __forceinline__ unsigned pack2bf(float lo, float hi) {
  return (unsigned)f2bf(lo) | ((unsigned)f2bf(hi) << 16);
}
#endif

__device__ __forceinline__ v8f wmma_bf16(v16bf a, v16bf b, v8f c) {
  return __builtin_amdgcn_wmma_f32_16x16x32_bf16(
      /*neg_a=*/false, a, /*neg_b=*/false, b,
      /*c_mod=*/(short)0, c, /*reuse_a=*/false, /*reuse_b=*/false);
}

// A-fragment (16-bit, 16x32, MxK): lane L: m = L%16, half = L/16.
// elem i -> k = kbase + (i<8 ? i : i+8) + 8*half   (ISA 16-bit A 16x32 table)
template<int LD>
__device__ __forceinline__ v16bf load_a_frag(const unsigned short* tile, int lane,
                                             int kbase) {
  const int m = lane & 15, half = lane >> 4;
  BFrag f;
#pragma unroll
  for (int i = 0; i < 16; ++i) {
    const int k = kbase + (i < 8 ? i : i + 8) + 8 * half;
    f.u[i] = tile[m * LD + k];
  }
  return f.v;
}

// B-fragment (16-bit, 32x16, KxN) read from an N-major tile: tile[n][k].
// lane L: n = nbase + L%16 ; k = kbase + 8*(L/16) + (i<8 ? i : i+8)  (ISA table order)
template<int LD>
__device__ __forceinline__ v16bf load_b_frag_nk(const unsigned short* tile,
                                                int nbase, int lane, int kbase) {
  const int n  = nbase + (lane & 15);
  const int kh = kbase + ((lane >> 4) << 3);
  BFrag f;
#pragma unroll
  for (int i = 0; i < 16; ++i) f.u[i] = tile[n * LD + kh + (i < 8 ? i : i + 8)];
  return f.v;
}

// ---------------------------------------------------------------------------
// Kernel 1: fused QKV GEMM + RMSNorm + RoPE.
// grid = (BS/16, 32): y-slot 0..15 = Q heads, 16..23 = K heads, 24..31 = V heads.
// Block tile: 16 tokens x 128 dims (one full head); 64-wide K slabs, software
// pipelined (next slab's global loads issued while current slab's WMMAs run).
// V is written TRANSPOSED ([b][hkv][d][s]) so flash attention gets N-major V.
// ---------------------------------------------------------------------------
__global__ void __launch_bounds__(256)
qkv_rope_kernel(const float* __restrict__ X,
                const float* __restrict__ cosT, const float* __restrict__ sinT,
                const float* __restrict__ Wq, const float* __restrict__ Wk,
                const float* __restrict__ Wv,
                const float* __restrict__ qw, const float* __restrict__ kw,
                unsigned short* __restrict__ Qbf,
                unsigned short* __restrict__ Kbf,
                unsigned short* __restrict__ VbfT) {
  __shared__ alignas(16) unsigned short As[16 * 64];
  __shared__ alignas(16) unsigned short Bs[128 * 64];
  __shared__ float          Cs[16 * 128];
  __shared__ float          rstd[16];

  const int tid  = threadIdx.x;
  const int lane = tid & 31;
  const int w    = tid >> 5;
  const int t0   = blockIdx.x * 16;
  const int slot = blockIdx.y;
  int kind, head;
  if (slot < 16)      { kind = 0; head = slot; }
  else if (slot < 24) { kind = 1; head = slot - 16; }
  else                { kind = 2; head = slot - 24; }
  const float* Wp  = (kind == 0) ? Wq : ((kind == 1) ? Wk : Wv);
  const int    wr0 = head * Dd;

  const int ar  = tid >> 4;            // A: row 0..15
  const int ac4 = (tid & 15) * 4;      // A: col (x4 floats)

  float4 aReg;
  float4 bReg[8];
  auto loadSlab = [&](int k0) {
    aReg = *(const float4*)&X[(size_t)(t0 + ar) * Hh + k0 + ac4];
#pragma unroll
    for (int j = 0; j < 8; ++j) {
      const int s  = tid + j * 256;    // 0..2047 float4 slots
      const int r  = s >> 4;           // 0..127
      const int c4 = (s & 15) * 4;
      bReg[j] = *(const float4*)&Wp[(size_t)(wr0 + r) * Hh + k0 + c4];
    }
  };
  auto storeSlab = [&]() {
    *(uint2*)&As[ar * 64 + ac4] =
        make_uint2(pack2bf(aReg.x, aReg.y), pack2bf(aReg.z, aReg.w));
#pragma unroll
    for (int j = 0; j < 8; ++j) {
      const int s  = tid + j * 256;
      const int r  = s >> 4;
      const int c4 = (s & 15) * 4;
      *(uint2*)&Bs[r * 64 + c4] =
          make_uint2(pack2bf(bReg[j].x, bReg[j].y), pack2bf(bReg[j].z, bReg[j].w));
    }
  };

  v8f acc = {0.f, 0.f, 0.f, 0.f, 0.f, 0.f, 0.f, 0.f};

  loadSlab(0);
  for (int k0 = 0; k0 < Hh; k0 += 64) {
    storeSlab();
    __syncthreads();
    if (k0 + 64 < Hh) loadSlab(k0 + 64);   // overlap with WMMAs below
#pragma unroll
    for (int kc = 0; kc < 2; ++kc) {
      const v16bf a  = load_a_frag<64>(As, lane, kc * 32);
      const v16bf bf = load_b_frag_nk<64>(Bs, w * 16, lane, kc * 32);
      acc = wmma_bf16(a, bf, acc);
    }
    __syncthreads();
  }

  // C layout: VGPR r -> row r + 8*(lane/16), col = lane%16 (within wave's 16 cols)
  {
    const int half = lane >> 4, n16 = lane & 15;
#pragma unroll
    for (int r = 0; r < 8; ++r)
      Cs[(r + 8 * half) * 128 + w * 16 + n16] = acc[r];
  }
  __syncthreads();

  if (kind == 2) {  // V: bf16 store TRANSPOSED [b][hkv][d][s], packed 2 tokens
#pragma unroll
    for (int p = 0; p < 4; ++p) {
      const int d  = (tid >> 3) + p * 32;        // 0..127
      const int sp = (tid & 7) * 2;              // token pair in tile
      const int t = t0 + sp, b = t / Ss, s = t % Ss;
      const unsigned pk = pack2bf(Cs[sp * 128 + d], Cs[(sp + 1) * 128 + d]);
      *(unsigned*)&VbfT[(((size_t)b * NKV + head) * Dd + d) * Ss + s] = pk;
    }
  } else {          // Q/K: RMSNorm over D=128 then RoPE
    const int r = tid >> 4, g = tid & 15;
    float ss = 0.f;
#pragma unroll
    for (int j = 0; j < 8; ++j) {
      const float x = Cs[r * 128 + g * 8 + j];
      ss += x * x;
    }
#pragma unroll
    for (int m = 1; m < 16; m <<= 1) ss += __shfl_xor(ss, m, 16);
    if (g == 0) rstd[r] = rsqrtf(ss * (1.0f / 128.0f) + EPS);
    __syncthreads();

    const float* nw = (kind == 0) ? qw : kw;
    unsigned short* dst = (kind == 0) ? Qbf : Kbf;
    const int nheads = (kind == 0) ? NH : NKV;
#pragma unroll
    for (int j = 0; j < 4; ++j) {            // packed pairs: 4 b32 stores/thread
      const int i  = tid + j * 256;          // 0..1023 pair slots
      const int rr = i >> 6;                 // token row 0..15
      const int d0 = (i & 63) * 2;           // even d
      const int t = t0 + rr, b = t / Ss, s = t % Ss;
      const float rs = rstd[rr];
      float y[2];
#pragma unroll
      for (int e = 0; e < 2; ++e) {
        const int d  = d0 + e;
        const int dp = d ^ 64;               // rotate_half partner
        const float xn = Cs[rr * 128 + d]  * rs * nw[d];
        const float xo = Cs[rr * 128 + dp] * rs * nw[dp];
        const float cv = cosT[((size_t)b * Ss + s) * Dd + d];
        const float sv = sinT[((size_t)b * Ss + s) * Dd + d];
        const float rot = (d < 64) ? -xo : xo;
        y[e] = xn * cv + rot * sv;
      }
      *(unsigned*)&dst[(((size_t)b * nheads + head) * Ss + s) * Dd + d0] =
          pack2bf(y[0], y[1]);
    }
  }
}

// ---------------------------------------------------------------------------
// Kernel 2: causal flash attention, bf16 WMMA, f32 online softmax.
// grid = (S/128, NH, B); 8 waves; wave w owns q rows [qb*128+16w, +16).
// K (row-major) and V^T (d-major) tiles are DOUBLE-BUFFERED in LDS and staged
// with GLOBAL_LOAD_ASYNC_TO_LDS_B128: next tile's DMA overlaps current WMMAs.
// ---------------------------------------------------------------------------
__global__ void __launch_bounds__(256)
flash_attn_kernel(const unsigned short* __restrict__ Qbf,
                  const unsigned short* __restrict__ Kbf,
                  const unsigned short* __restrict__ VbfT,
                  unsigned short* __restrict__ Ctx) {
  __shared__ alignas(16) unsigned short Kt [2][32 * 128]; // [kv][d]
  __shared__ alignas(16) unsigned short VtT[2][128 * 32]; // [d][kv]
  __shared__ alignas(16) unsigned short Ps[8 * 16 * 32];  // per-wave P staging

  const int tid  = threadIdx.x, lane = tid & 31, w = tid >> 5;
  const int qb   = blockIdx.x, h = blockIdx.y, b = blockIdx.z;
  const int hkv  = h >> 1;                     // NH/NKV = 2
  const int q0   = qb * 128 + w * 16;
  const int half = lane >> 4, n16 = lane & 15;

  // Preload Q strip as 4 A-fragments; A layout = two contiguous 16B runs.
  v16bf qf[4];
  {
    const unsigned short* Qp = Qbf + (((size_t)b * NH + h) * Ss + q0) * Dd;
#pragma unroll
    for (int c = 0; c < 4; ++c) {
      BFrag f;
      f.q[0] = *(const uint4*)&Qp[(size_t)n16 * Dd + c * 32 + 8 * half];      // i=0..7
      f.q[1] = *(const uint4*)&Qp[(size_t)n16 * Dd + c * 32 + 16 + 8 * half]; // i=8..15
      qf[c] = f.v;
    }
  }

  const v8f vz = {0.f, 0.f, 0.f, 0.f, 0.f, 0.f, 0.f, 0.f};
  v8f o[8];
#pragma unroll
  for (int i = 0; i < 8; ++i) o[i] = vz;
  float m_[8], l_[8];
#pragma unroll
  for (int r = 0; r < 8; ++r) { m_[r] = -1e30f; l_[r] = 0.f; }

  const float scale    = 0.088388347648318447f;   // 1/sqrt(128)
  const int   kv_end   = qb * 128 + 128;          // causal bound for the block
  const int   wave_qmx = q0 + 15;

  const unsigned short* Kbase  = Kbf  + (((size_t)b * NKV + hkv) * Ss) * Dd;
  const unsigned short* VTbase = VbfT + (((size_t)b * NKV + hkv) * Dd) * Ss;

  // Async-stage one 32-kv step (K: 8KB row-major, V^T: 8KB d-major) into buf i.
  auto stageKV = [&](int bufi, int kv0) {
    const unsigned lK = (unsigned)(uintptr_t)&Kt[bufi][0] + (unsigned)tid * 16u;
    const unsigned long long gK =
        (unsigned long long)(uintptr_t)(Kbase + (size_t)kv0 * Dd) + (unsigned)tid * 16u;
    asm volatile("global_load_async_to_lds_b128 %0, %1, off"
                 :: "v"(lK), "v"(gK) : "memory");
    asm volatile("global_load_async_to_lds_b128 %0, %1, off offset:4096"
                 :: "v"(lK), "v"(gK) : "memory");
#pragma unroll
    for (int hh = 0; hh < 2; ++hh) {
      const int slot = tid + hh * 256;           // 0..511
      const int row = slot >> 2, chunk = slot & 3;  // d row, 16B chunk of 32 kv
      const unsigned lV = (unsigned)(uintptr_t)&VtT[bufi][0]
                        + (unsigned)(row * 64 + chunk * 16);
      const unsigned long long gV =
          (unsigned long long)(uintptr_t)(VTbase + ((size_t)row * Ss + kv0 + chunk * 8));
      asm volatile("global_load_async_to_lds_b128 %0, %1, off"
                   :: "v"(lV), "v"(gV) : "memory");
    }
  };

  stageKV(0, 0);
  asm volatile("s_wait_asynccnt 0x0" ::: "memory");
  __syncthreads();

  const int nIter = kv_end / 32;
  for (int it = 0; it < nIter; ++it) {
    const int kv0 = it * 32;
    const int cur = it & 1;
    if (it + 1 < nIter) stageKV(cur ^ 1, kv0 + 32);   // DMA overlaps compute

    if (kv0 <= wave_qmx) {                 // wave-uniform: EXEC stays all-1s
      const unsigned short* Kc = &Kt[cur][0];
      const unsigned short* Vc = &VtT[cur][0];

      // ---- scores: two 16x16 tiles over kv cols [kv0, kv0+32) ----
      v8f sc0 = vz, sc1 = vz;
#pragma unroll
      for (int c = 0; c < 4; ++c) {
        sc0 = wmma_bf16(qf[c], load_b_frag_nk<128>(Kc,  0, lane, c * 32), sc0);
        sc1 = wmma_bf16(qf[c], load_b_frag_nk<128>(Kc, 16, lane, c * 32), sc1);
      }

      // ---- online softmax (row reductions across 16-lane halves) ----
      float alpha[8], p0[8], p1[8];
#pragma unroll
      for (int r = 0; r < 8; ++r) {
        const int qrow = q0 + r + 8 * half;
        float v0 = sc0[r] * scale;
        float v1 = sc1[r] * scale;
        if (kv0 + n16      > qrow) v0 = -1e30f;
        if (kv0 + 16 + n16 > qrow) v1 = -1e30f;
        float rm = fmaxf(v0, v1);
#pragma unroll
        for (int mm = 1; mm < 16; mm <<= 1) rm = fmaxf(rm, __shfl_xor(rm, mm, 16));
        const float mn = fmaxf(m_[r], rm);
        alpha[r] = __expf(m_[r] - mn);
        const float e0 = __expf(v0 - mn);
        const float e1 = __expf(v1 - mn);
        float rs = e0 + e1;
#pragma unroll
        for (int mm = 1; mm < 16; mm <<= 1) rs += __shfl_xor(rs, mm, 16);
        l_[r] = l_[r] * alpha[r] + rs;
        m_[r] = mn;
        p0[r] = e0; p1[r] = e1;
      }
#pragma unroll
      for (int dch = 0; dch < 8; ++dch)
#pragma unroll
        for (int r = 0; r < 8; ++r) o[dch][r] *= alpha[r];

      // ---- C-layout -> A-layout transpose of P through per-wave LDS ----
      unsigned short* Pw = Ps + w * (16 * 32);
#pragma unroll
      for (int r = 0; r < 8; ++r) {
        Pw[(r + 8 * half) * 32 + n16]      = f2bf(p0[r]);
        Pw[(r + 8 * half) * 32 + 16 + n16] = f2bf(p1[r]);
      }
      asm volatile("s_wait_dscnt 0x0" ::: "memory");   // wave-local LDS RAW
      const v16bf pa = load_a_frag<32>(Pw, lane, 0);

      // ---- O += P @ V : V^T tile is N-major -> 2x ds_load_b128 per frag ----
#pragma unroll
      for (int dch = 0; dch < 8; ++dch)
        o[dch] = wmma_bf16(pa, load_b_frag_nk<32>(Vc, dch * 16, lane, 0), o[dch]);
    }
    asm volatile("s_wait_asynccnt 0x0" ::: "memory");  // next tile landed
    __syncthreads();
  }

  // ---- epilogue: normalize and store context bf16 [b][s][h*D + d] ----
#pragma unroll
  for (int r = 0; r < 8; ++r) {
    const float inv = 1.f / l_[r];
    const int   s   = q0 + r + 8 * half;
    const size_t base = ((size_t)b * Ss + s) * (NH * Dd) + (size_t)h * Dd;
#pragma unroll
    for (int dch = 0; dch < 8; ++dch)
      Ctx[base + dch * 16 + n16] = f2bf(o[dch][r] * inv);
  }
}

// ---------------------------------------------------------------------------
// Kernel 3: output projection  out = Ctx(4096x2048 bf16) @ Wo^T  -> f32.
// grid = (BS/16, H/128); block tile 16 x 128; 64-wide slabs, software pipelined.
// ---------------------------------------------------------------------------
__global__ void __launch_bounds__(256)
oproj_kernel(const unsigned short* __restrict__ Ctx,
             const float* __restrict__ Wo,
             float* __restrict__ Out) {
  __shared__ alignas(16) unsigned short As[16 * 64];
  __shared__ alignas(16) unsigned short Bs[128 * 64];

  const int tid = threadIdx.x, lane = tid & 31, w = tid >> 5;
  const int t0 = blockIdx.x * 16;
  const int c0 = blockIdx.y * 128;
  constexpr int K = NH * Dd;     // 2048

  const int ar  = tid >> 4;
  const int ac4 = (tid & 15) * 4;

  uint2  aReg;
  float4 bReg[8];
  auto loadSlab = [&](int k0) {
    aReg = *(const uint2*)&Ctx[(size_t)(t0 + ar) * K + k0 + ac4];
#pragma unroll
    for (int j = 0; j < 8; ++j) {
      const int s  = tid + j * 256;
      const int r  = s >> 4;
      const int c4 = (s & 15) * 4;
      bReg[j] = *(const float4*)&Wo[(size_t)(c0 + r) * K + k0 + c4];
    }
  };
  auto storeSlab = [&]() {
    *(uint2*)&As[ar * 64 + ac4] = aReg;
#pragma unroll
    for (int j = 0; j < 8; ++j) {
      const int s  = tid + j * 256;
      const int r  = s >> 4;
      const int c4 = (s & 15) * 4;
      *(uint2*)&Bs[r * 64 + c4] =
          make_uint2(pack2bf(bReg[j].x, bReg[j].y), pack2bf(bReg[j].z, bReg[j].w));
    }
  };

  v8f acc = {0.f, 0.f, 0.f, 0.f, 0.f, 0.f, 0.f, 0.f};

  loadSlab(0);
  for (int k0 = 0; k0 < K; k0 += 64) {
    storeSlab();
    __syncthreads();
    if (k0 + 64 < K) loadSlab(k0 + 64);    // overlap with WMMAs below
#pragma unroll
    for (int kc = 0; kc < 2; ++kc) {
      const v16bf a  = load_a_frag<64>(As, lane, kc * 32);
      const v16bf bf = load_b_frag_nk<64>(Bs, w * 16, lane, kc * 32);
      acc = wmma_bf16(a, bf, acc);
    }
    __syncthreads();
  }

  const int half = lane >> 4, n16 = lane & 15;
#pragma unroll
  for (int r = 0; r < 8; ++r)
    Out[(size_t)(t0 + r + 8 * half) * Hh + c0 + w * 16 + n16] = acc[r];
}

// ---------------------------------------------------------------------------
// Host entry
// ---------------------------------------------------------------------------
extern "C" void kernel_launch(void* const* d_in, const int* in_sizes, int n_in,
                              void* d_out, int out_size, void* d_ws, size_t ws_size,
                              hipStream_t stream) {
  const float* X    = (const float*)d_in[0];
  const float* cosT = (const float*)d_in[1];
  const float* sinT = (const float*)d_in[2];
  const float* Wq   = (const float*)d_in[3];
  const float* Wk   = (const float*)d_in[4];
  const float* Wv   = (const float*)d_in[5];
  const float* Wo   = (const float*)d_in[6];
  const float* qw   = (const float*)d_in[7];
  const float* kw   = (const float*)d_in[8];
  float* Out = (float*)d_out;

  // Workspace layout (bf16 buffers)
  char* ws = (char*)d_ws;
  const size_t szQ = (size_t)Bb * NH  * Ss * Dd * sizeof(unsigned short); // 16 MB
  const size_t szK = (size_t)Bb * NKV * Ss * Dd * sizeof(unsigned short); //  8 MB
  const size_t szV = szK;                                                 //  8 MB
  unsigned short* Qbf  = (unsigned short*)(ws);
  unsigned short* Kbf  = (unsigned short*)(ws + szQ);
  unsigned short* VbfT = (unsigned short*)(ws + szQ + szK);
  unsigned short* Ctx  = (unsigned short*)(ws + szQ + szK + szV);         // 16 MB

  const int BS = Bb * Ss;

  qkv_rope_kernel<<<dim3(BS / 16, NH + NKV + NKV), 256, 0, stream>>>(
      X, cosT, sinT, Wq, Wk, Wv, qw, kw, Qbf, Kbf, VbfT);

  flash_attn_kernel<<<dim3(Ss / 128, NH, Bb), 256, 0, stream>>>(
      Qbf, Kbf, VbfT, Ctx);

  oproj_kernel<<<dim3(BS / 16, Hh / 128), 256, 0, stream>>>(Ctx, Wo, Out);
}
